// SFE_22368189678214
// MI455X (gfx1250) — matrix operand order repaired and checked by reading, not compiled, not run on hardware
//
#include <hip/hip_runtime.h>

typedef __bf16 bf16;
typedef __attribute__((ext_vector_type(16))) __bf16 v16bf;
typedef __attribute__((ext_vector_type(8)))  float  v8f;

#define BDIM 4
#define CDIM 64
#define NDIM 4096
#define C2DIM 128
#define CRDIM 8

// ---- WMMA wrapper: D = A(16x32 bf16) * B(32x16 bf16) + C(16x16 f32) ----
__device__ __forceinline__ v8f wmma_bf16(v16bf a, v16bf b, v8f c) {
  return __builtin_amdgcn_wmma_f32_16x16x32_bf16(
      /*neg_a=*/false, a, /*neg_b=*/false, b,
      /*c_mod=*/(short)0, c, /*reuse_a=*/false, /*reuse_b=*/false);
}

// A fragment (16x32, M x K) from row-major [m][k] storage.
// Lane l: m = l&15, h = l>>4. VGPR0-3 hold K [8h,8h+8), VGPR4-7 hold K [16+8h,24+8h).
__device__ __forceinline__ v16bf frag_a(const bf16* base, int ld, int k0, int lane) {
  int m = lane & 15, h = lane >> 4;
  const bf16* p = base + (size_t)m * ld + k0;
  union { v16bf v; uint4 q[2]; } u;
  u.q[0] = *(const uint4*)(p + 8 * h);
  u.q[1] = *(const uint4*)(p + 16 + 8 * h);
  return u.v;
}

// B fragment (32x16, K x N) from B^T stored row-major [n][k].
// Lane l: n = l&15, h = l>>4. VGPR0-3 hold K [8h,8h+8), VGPR4-7 hold K [16+8h,24+8h).
__device__ __forceinline__ v16bf frag_b(const bf16* base, int ld, int k0, int lane) {
  int n = lane & 15, h = lane >> 4;
  const bf16* p = base + (size_t)n * ld + k0;
  union { v16bf v; uint4 q[2]; } u;
  u.q[0] = *(const uint4*)(p + 8 * h);
  u.q[1] = *(const uint4*)(p + 16 + 8 * h);
  return u.v;
}

// ---- DPP16 ROW_XMASK cross-lane reductions within a row of 16 (VALU-only) ----
template <int CTRL>
__device__ __forceinline__ float dpp_xor(float x) {
  int xi = __builtin_bit_cast(int, x);
  int r = __builtin_amdgcn_update_dpp(xi, xi, CTRL, 0xF, 0xF, false);
  return __builtin_bit_cast(float, r);
}
__device__ __forceinline__ float row16_max(float x) {
  x = fmaxf(x, dpp_xor<0x161>(x));  // ROW_XMASK:1
  x = fmaxf(x, dpp_xor<0x162>(x));  // ROW_XMASK:2
  x = fmaxf(x, dpp_xor<0x164>(x));  // ROW_XMASK:4
  x = fmaxf(x, dpp_xor<0x168>(x));  // ROW_XMASK:8
  return x;
}
__device__ __forceinline__ float row16_sum(float x) {
  x += dpp_xor<0x161>(x);
  x += dpp_xor<0x162>(x);
  x += dpp_xor<0x164>(x);
  x += dpp_xor<0x168>(x);
  return x;
}

// ---- 1) prep: bf16 copies + transposes via LDS tiles ----
// grid: B*(C/16)*(N/16), block (16,16)
__global__ __launch_bounds__(256) void prep_kernel(const float* __restrict__ x1,
                                                   const float* __restrict__ x2,
                                                   bf16* __restrict__ A1c, bf16* __restrict__ A2c,
                                                   bf16* __restrict__ Qb,  bf16* __restrict__ Kb) {
  __shared__ float t1[16][17];
  __shared__ float t2[16][17];
  int bid = blockIdx.x;
  int nt = bid % (NDIM / 16); bid /= (NDIM / 16);
  int ct = bid % (CDIM / 16);
  int b  = bid / (CDIM / 16);
  int tx = threadIdx.x, ty = threadIdx.y;
  size_t idx = ((size_t)b * CDIM + ct * 16 + ty) * NDIM + nt * 16 + tx;
  float v1 = x1[idx], v2 = x2[idx];
  A1c[idx] = (bf16)v1;
  A2c[idx] = (bf16)v2;
  t1[ty][tx] = v1;
  t2[ty][tx] = v2;
  __syncthreads();
  size_t tidx = ((size_t)b * NDIM + nt * 16 + ty) * CDIM + ct * 16 + tx;
  Qb[tidx] = (bf16)t1[tx][ty];
  Kb[tidx] = (bf16)t2[tx][ty];
}

// ---- 2) Vt[b][d][n] = sum_c a2[b][c][n] * weight[c][d]  (V stored column-major) ----
// grid: B*C*(N/256), block 256
__global__ __launch_bounds__(256) void support_kernel(const float* __restrict__ x2,
                                                      const float* __restrict__ w,
                                                      bf16* __restrict__ Vt) {
  int bid = blockIdx.x;
  int nb = bid % (NDIM / 256); bid /= (NDIM / 256);
  int d = bid % CDIM;
  int b = bid / CDIM;
  int n = nb * 256 + threadIdx.x;
  float acc = 0.f;
#pragma unroll 8
  for (int c = 0; c < CDIM; c++)
    acc += x2[((size_t)b * CDIM + c) * NDIM + n] * w[c * CDIM + d];
  Vt[((size_t)b * CDIM + d) * NDIM + n] = (bf16)acc;
}

// ---- 3) flash attention: y = softmax(Q K^T) V, write into z[:,0:C,:] ----
// grid: B*(N/64), block 128 (4 waves, 16 rows each); 64-column tiles per iter
__global__ __launch_bounds__(128) void flash_kernel(const bf16* __restrict__ Qb,
                                                    const bf16* __restrict__ Kb,
                                                    const bf16* __restrict__ Vt,
                                                    float* __restrict__ z) {
  int b    = blockIdx.x / (NDIM / 64);
  int rb   = blockIdx.x % (NDIM / 64);
  int wave = threadIdx.x >> 5;
  int lane = threadIdx.x & 31;
  int hh   = lane >> 4;
  int row0 = rb * 64 + wave * 16;

  const bf16* Qbase = Qb + ((size_t)b * NDIM + row0) * CDIM;
  const bf16* Kbase = Kb + (size_t)b * NDIM * CDIM;
  const bf16* Vbase = Vt + (size_t)b * CDIM * NDIM;

  v16bf qa0 = frag_a(Qbase, CDIM, 0, lane);
  v16bf qa1 = frag_a(Qbase, CDIM, 32, lane);

  v8f o[4] = {{}, {}, {}, {}};
  float rmax[8], rsum[8];
#pragma unroll
  for (int v = 0; v < 8; v++) { rmax[v] = -3.0e38f; rsum[v] = 0.f; }

  __shared__ __align__(16) bf16 plds[4][16][64];

  for (int j = 0; j < NDIM; j += 64) {
    // ---- S = Q K^T for columns [j, j+64): 4 tiles of 16 ----
    v8f s[4];
#pragma unroll
    for (int t = 0; t < 4; t++) {
      const bf16* kb = Kbase + (size_t)(j + 16 * t) * CDIM;
      v8f acc = {};
      acc = wmma_bf16(qa0, frag_b(kb, CDIM, 0, lane), acc);
      acc = wmma_bf16(qa1, frag_b(kb, CDIM, 32, lane), acc);
      s[t] = acc;
    }

    // prefetch next K rows / V row-segments (global_prefetch_b8)
    __builtin_prefetch(Kbase + (size_t)(j + 64 + lane) * CDIM, 0, 3);
    __builtin_prefetch(Kbase + (size_t)(j + 96 + lane) * CDIM, 0, 3);
    __builtin_prefetch(Vbase + (size_t)lane * NDIM + j + 64, 0, 3);
    __builtin_prefetch(Vbase + (size_t)(32 + lane) * NDIM + j + 64, 0, 3);

    // ---- online softmax per row (row m = v + 8*hh, spread across 16 lanes) ----
    float sc[8];
#pragma unroll
    for (int v = 0; v < 8; v++) {
      float t = fmaxf(fmaxf(s[0][v], s[1][v]), fmaxf(s[2][v], s[3][v]));
      t = row16_max(t);
      float nm = fmaxf(rmax[v], t);
      sc[v] = __expf(rmax[v] - nm);
      rmax[v] = nm;
      float p0 = __expf(s[0][v] - nm);
      float p1 = __expf(s[1][v] - nm);
      float p2 = __expf(s[2][v] - nm);
      float p3 = __expf(s[3][v] - nm);
      s[0][v] = p0; s[1][v] = p1; s[2][v] = p2; s[3][v] = p3;
      float rs = row16_sum((p0 + p1) + (p2 + p3));
      rsum[v] = rsum[v] * sc[v] + rs;
    }
#pragma unroll
    for (int v = 0; v < 8; v++) {
      o[0][v] *= sc[v]; o[1][v] *= sc[v]; o[2][v] *= sc[v]; o[3][v] *= sc[v];
    }

    // ---- relayout P (C-layout f32) -> A-fragments (bf16) through LDS ----
    int kc = lane & 15;
#pragma unroll
    for (int v = 0; v < 8; v++) {
      int m = v + 8 * hh;
      plds[wave][m][kc]      = (bf16)s[0][v];
      plds[wave][m][kc + 16] = (bf16)s[1][v];
      plds[wave][m][kc + 32] = (bf16)s[2][v];
      plds[wave][m][kc + 48] = (bf16)s[3][v];
    }
    asm volatile("s_wait_dscnt 0" ::: "memory");
    v16bf pa0 = frag_a(&plds[wave][0][0], 64, 0, lane);
    v16bf pa1 = frag_a(&plds[wave][0][0], 64, 32, lane);
    asm volatile("s_wait_dscnt 0" ::: "memory");

    // ---- O += P * V   (V rows [j, j+64), 4 column tiles) ----
#pragma unroll
    for (int ct = 0; ct < 4; ct++) {
      const bf16* vb = Vbase + (size_t)(16 * ct) * NDIM + j;
      o[ct] = wmma_bf16(pa0, frag_b(vb, NDIM, 0, lane), o[ct]);
      o[ct] = wmma_bf16(pa1, frag_b(vb, NDIM, 32, lane), o[ct]);
    }
  }

  // normalize and scatter y into z[b][d][n] (channels 0..C)
  int nc = lane & 15;
#pragma unroll
  for (int v = 0; v < 8; v++) {
    int row = row0 + v + 8 * hh;
    float inv = 1.f / rsum[v];
    z[((size_t)b * C2DIM + (nc +  0)) * NDIM + row] = o[0][v] * inv;
    z[((size_t)b * C2DIM + (nc + 16)) * NDIM + row] = o[1][v] * inv;
    z[((size_t)b * C2DIM + (nc + 32)) * NDIM + row] = o[2][v] * inv;
    z[((size_t)b * C2DIM + (nc + 48)) * NDIM + row] = o[3][v] * inv;
  }
}

// ---- 4) c_x[b] = a1 @ a2^T  (64x64, K=4096) via WMMA ----
// grid: B, block 128
__global__ __launch_bounds__(128) void cx_kernel(const bf16* __restrict__ A1c,
                                                 const bf16* __restrict__ A2c,
                                                 float* __restrict__ cx) {
  int b = blockIdx.x;
  int wave = threadIdx.x >> 5;
  int lane = threadIdx.x & 31;
  int hh = lane >> 4;
  const bf16* Abase = A1c + ((size_t)b * CDIM + wave * 16) * NDIM;
  const bf16* Bbase = A2c + (size_t)b * CDIM * NDIM;
  for (int dt = 0; dt < 4; dt++) {
    v8f acc = {};
    for (int k = 0; k < NDIM; k += 32)
      acc = wmma_bf16(frag_a(Abase, NDIM, k, lane),
                      frag_b(Bbase + (size_t)dt * 16 * NDIM, NDIM, k, lane), acc);
#pragma unroll
    for (int v = 0; v < 8; v++)
      cx[((size_t)b * CDIM + wave * 16 + v + 8 * hh) * CDIM + dt * 16 + (lane & 15)] = acc[v];
  }
}

// ---- 5) support2 = c_x @ weight2 (tiny), bf16 row-major ----
// grid: B, block 256
__global__ __launch_bounds__(256) void s2_kernel(const float* __restrict__ cx,
                                                 const float* __restrict__ w2,
                                                 bf16* __restrict__ s2b) {
  int b = blockIdx.x;
  for (int i = threadIdx.x; i < CDIM * CDIM; i += blockDim.x) {
    int c = i >> 6, e = i & 63;
    float acc = 0.f;
#pragma unroll 8
    for (int d = 0; d < CDIM; d++)
      acc += cx[((size_t)b * CDIM + c) * CDIM + d] * w2[d * CDIM + e];
    s2b[((size_t)b * CDIM + c) * CDIM + e] = (bf16)acc;
  }
}

// ---- 6) y2 = support2 @ a1 (B operand straight from Qb = a1^T), into z[:,C:2C,:] ----
// grid: B*(N/64), block 128
__global__ __launch_bounds__(128) void y2_kernel(const bf16* __restrict__ s2b,
                                                 const bf16* __restrict__ Qb,
                                                 float* __restrict__ z) {
  int b  = blockIdx.x / (NDIM / 64);
  int nb = blockIdx.x % (NDIM / 64);
  int wave = threadIdx.x >> 5;
  int lane = threadIdx.x & 31;
  int hh = lane >> 4;
  const bf16* Abase = s2b + ((size_t)b * CDIM + wave * 16) * CDIM;
  const bf16* Bbase = Qb + ((size_t)b * NDIM + nb * 64) * CDIM;
  v16bf a0 = frag_a(Abase, CDIM, 0, lane);
  v16bf a1f = frag_a(Abase, CDIM, 32, lane);
  for (int ct = 0; ct < 4; ct++) {
    v8f acc = {};
    acc = wmma_bf16(a0,  frag_b(Bbase + (size_t)ct * 16 * CDIM, CDIM, 0, lane), acc);
    acc = wmma_bf16(a1f, frag_b(Bbase + (size_t)ct * 16 * CDIM, CDIM, 32, lane), acc);
#pragma unroll
    for (int v = 0; v < 8; v++)
      z[((size_t)b * C2DIM + CDIM + wave * 16 + v + 8 * hh) * NDIM + nb * 64 + ct * 16 + (lane & 15)] = acc[v];
  }
}

// ---- 7) global max pool over spatial ----
// grid: B*C2, block 256
__global__ __launch_bounds__(256) void pool_kernel(const float* __restrict__ z,
                                                   float* __restrict__ mp) {
  __shared__ float red[256];
  size_t e = blockIdx.x;
  float m = -3.0e38f;
  for (int n = threadIdx.x; n < NDIM; n += 256) m = fmaxf(m, z[e * NDIM + n]);
  red[threadIdx.x] = m;
  __syncthreads();
  for (int s = 128; s > 0; s >>= 1) {
    if (threadIdx.x < s) red[threadIdx.x] = fmaxf(red[threadIdx.x], red[threadIdx.x + s]);
    __syncthreads();
  }
  if (threadIdx.x == 0) mp[e] = red[0];
}

// ---- 8) SE MLP: sv = 1 + sigmoid(fc2(relu(fc1(mp)))) ----
// grid: B, block 128
__global__ __launch_bounds__(128) void se_kernel(const float* __restrict__ mp,
                                                 const float* __restrict__ fc1,
                                                 const float* __restrict__ fc2,
                                                 float* __restrict__ sv) {
  int b = blockIdx.x;
  __shared__ float h[CRDIM];
  if (threadIdx.x < CRDIM) {
    float acc = 0.f;
    for (int e = 0; e < C2DIM; e++) acc += mp[b * C2DIM + e] * fc1[threadIdx.x * C2DIM + e];
    h[threadIdx.x] = fmaxf(acc, 0.f);
  }
  __syncthreads();
  int e = threadIdx.x;
  float acc = 0.f;
#pragma unroll
  for (int r = 0; r < CRDIM; r++) acc += h[r] * fc2[e * CRDIM + r];
  sv[b * C2DIM + e] = 1.f + 1.f / (1.f + __expf(-acc));
}

// ---- 9) final 1x1 conv: out[b][o][n] = sum_e conv_w[o][e]*sv[b][e]*z[b][e][n] + cb[o] ----
// grid: B*C*(N/256), block 256
__global__ __launch_bounds__(256) void conv_kernel(const float* __restrict__ z,
                                                   const float* __restrict__ sv,
                                                   const float* __restrict__ cw,
                                                   const float* __restrict__ cb,
                                                   float* __restrict__ out) {
  int bid = blockIdx.x;
  int nb = bid % (NDIM / 256); bid /= (NDIM / 256);
  int o = bid % CDIM;
  int b = bid / CDIM;
  int n = nb * 256 + threadIdx.x;
  float acc = cb[o];
#pragma unroll 8
  for (int e = 0; e < C2DIM; e++)
    acc += cw[o * C2DIM + e] * sv[b * C2DIM + e] * z[((size_t)b * C2DIM + e) * NDIM + n];
  out[((size_t)b * CDIM + o) * NDIM + n] = acc;
}

extern "C" void kernel_launch(void* const* d_in, const int* in_sizes, int n_in,
                              void* d_out, int out_size, void* d_ws, size_t ws_size,
                              hipStream_t stream) {
  const float* x1   = (const float*)d_in[0];
  const float* x2   = (const float*)d_in[1];
  const float* wgt  = (const float*)d_in[2];
  const float* wgt2 = (const float*)d_in[3];
  const float* fc1  = (const float*)d_in[4];
  const float* fc2  = (const float*)d_in[5];
  const float* cw   = (const float*)d_in[6];
  const float* cb   = (const float*)d_in[7];
  float* out = (float*)d_out;

  constexpr size_t NC = (size_t)BDIM * NDIM * CDIM;  // 1M elems
  bf16* Qb  = (bf16*)d_ws;
  bf16* Kb  = Qb + NC;
  bf16* Vt  = Kb + NC;
  bf16* A1c = Vt + NC;
  bf16* A2c = A1c + NC;
  float* z  = (float*)(A2c + NC);                    // B*2C*N f32 (8MB)
  float* cx = z + (size_t)BDIM * C2DIM * NDIM;
  float* mp = cx + (size_t)BDIM * CDIM * CDIM;
  float* sv = mp + (size_t)BDIM * C2DIM;
  bf16* s2b = (bf16*)(sv + (size_t)BDIM * C2DIM);

  prep_kernel<<<BDIM * (CDIM / 16) * (NDIM / 16), dim3(16, 16), 0, stream>>>(x1, x2, A1c, A2c, Qb, Kb);
  support_kernel<<<BDIM * CDIM * (NDIM / 256), 256, 0, stream>>>(x2, wgt, Vt);
  flash_kernel<<<BDIM * (NDIM / 64), 128, 0, stream>>>(Qb, Kb, Vt, z);
  cx_kernel<<<BDIM, 128, 0, stream>>>(A1c, A2c, cx);
  s2_kernel<<<BDIM, 256, 0, stream>>>(cx, wgt2, s2b);
  y2_kernel<<<BDIM * (NDIM / 64), 128, 0, stream>>>(s2b, Qb, z);
  pool_kernel<<<BDIM * C2DIM, 256, 0, stream>>>(z, mp);
  se_kernel<<<BDIM, 128, 0, stream>>>(mp, fc1, fc2, sv);
  conv_kernel<<<BDIM * CDIM * (NDIM / 256), 256, 0, stream>>>(z, sv, cw, cb, out);
}
